// SwinQFormer_16535624090351
// MI455X (gfx1250) — matrix operand order repaired and checked by reading, not compiled, not run on hardware
//
#include <hip/hip_runtime.h>

#define HID 768
#define LN_EPS 1e-5f
#define QK_SCALE 0.03608439182435161f   // 1/sqrt(768)

typedef __attribute__((ext_vector_type(16))) __bf16 bf16x16;
typedef __attribute__((ext_vector_type(8)))  float  v8f;
typedef __attribute__((ext_vector_type(4)))  unsigned int u32x4;

union Frag { bf16x16 v; u32x4 q[2]; };

__device__ __forceinline__ unsigned short f2bf(float f) {
  unsigned u = __float_as_uint(f);
  u += 0x7FFFu + ((u >> 16) & 1u);        // round-to-nearest-even
  return (unsigned short)(u >> 16);
}
__device__ __forceinline__ float bf2f(unsigned short h) {
  return __uint_as_float(((unsigned)h) << 16);
}

// ---------------- f32 -> bf16 convert ----------------
__global__ __launch_bounds__(256) void k_cvt_bf16(const float* __restrict__ src,
                                                  unsigned short* __restrict__ dst, int n) {
  int i = blockIdx.x * 256 + threadIdx.x;
  if (i < n) dst[i] = f2bf(src[i]);
}

// ------- transpose+convert: src[K][N] f32 -> dst[N][K] bf16 -------
__global__ __launch_bounds__(256) void k_transpose_cvt(const float* __restrict__ src,
                                                       unsigned short* __restrict__ dst,
                                                       int K, int N) {
  int i = blockIdx.x * 256 + threadIdx.x;
  if (i < K * N) {
    int n = i / K, k = i - n * K;
    dst[i] = f2bf(src[(size_t)k * N + n]);
  }
}

// ---------------- bf16 WMMA GEMM:  C = A(MxK) * BT(NxK)^T + bias ----------------
// block = 256 threads = 8 waves; wave tile 16x64 (4 accumulators); block tile 32x256.
// Software-pipelined: fragments for iteration k+1 are loaded while iteration k's
// four WMMAs execute, so the compiler can issue partial s_wait_loadcnt instead of
// draining all loads before each WMMA.
template<bool OUT_BF16>
__global__ __launch_bounds__(256) void k_gemm_bf16(
    const unsigned short* __restrict__ A,   // Mpad x K bf16, row-major
    const unsigned short* __restrict__ BT,  // N x K bf16 (pre-transposed weights)
    const float* __restrict__ bias,         // N
    void* __restrict__ Cout,                // M x N
    int M, int K) {
  const int N    = gridDim.x * 256;
  const int wave = threadIdx.x >> 5;
  const int lane = threadIdx.x & 31;
  const int l15  = lane & 15;
  const int hi   = lane >> 4;
  const int m0   = blockIdx.y * 32 + (wave >> 2) * 16;
  const int n0   = blockIdx.x * 256 + (wave & 3) * 64;

  // A fragment: lane<16 -> row m0+l15, K {0..7, 16..23}; lane>=16 -> K {8..15, 24..31}
  const unsigned short* Ap = A + (size_t)(m0 + l15) * K + hi * 8;
  // B fragment (from BT, N-major): lane<16 -> K {0..7, 16..23}; lane>=16 -> K {8..15, 24..31}
  const unsigned short* Bp[4];
#pragma unroll
  for (int t = 0; t < 4; ++t)
    Bp[t] = BT + (size_t)(n0 + t * 16 + l15) * K + hi * 8;

  v8f acc[4] = {};

  // prologue: load k0 = 0 fragments
  Frag a0, b0[4];
  a0.q[0] = *(const u32x4*)(Ap);
  a0.q[1] = *(const u32x4*)(Ap + 16);
#pragma unroll
  for (int t = 0; t < 4; ++t) {
    b0[t].q[0] = *(const u32x4*)(Bp[t]);
    b0[t].q[1] = *(const u32x4*)(Bp[t] + 16);
  }

#pragma unroll 2
  for (int k0 = 0; k0 < K; k0 += 32) {
    // prefetch next iteration (wraps to 0 on the last trip: harmless dummy loads)
    int k1 = (k0 + 32 < K) ? (k0 + 32) : 0;
    Frag a1, b1[4];
    a1.q[0] = *(const u32x4*)(Ap + k1);
    a1.q[1] = *(const u32x4*)(Ap + k1 + 16);
#pragma unroll
    for (int t = 0; t < 4; ++t) {
      b1[t].q[0] = *(const u32x4*)(Bp[t] + k1);
      b1[t].q[1] = *(const u32x4*)(Bp[t] + k1 + 16);
    }
#pragma unroll
    for (int t = 0; t < 4; ++t)
      acc[t] = __builtin_amdgcn_wmma_f32_16x16x32_bf16(
          false, a0.v, false, b0[t].v, (short)0, acc[t], false, false);
    a0 = a1;
#pragma unroll
    for (int t = 0; t < 4; ++t) b0[t] = b1[t];
  }

#pragma unroll
  for (int t = 0; t < 4; ++t) {
    int col = n0 + t * 16 + l15;
    float bv = bias[col];
#pragma unroll
    for (int r = 0; r < 8; ++r) {
      int row = m0 + r + hi * 8;           // C/D layout: VGPR r, lane-half selects M+8
      if (row < M) {
        float val = acc[t][r] + bv;
        if (OUT_BF16)
          ((unsigned short*)Cout)[(size_t)row * N + col] = f2bf(val);
        else
          ((float*)Cout)[(size_t)row * N + col] = val;
      }
    }
  }
}

// ---------------- in-place LayerNorm over a 768-wide segment, bf16 storage ----------------
__global__ __launch_bounds__(256) void k_ln_bf16(unsigned short* __restrict__ data,
                                                 int rowStride, int segOff,
                                                 const float* __restrict__ g,
                                                 const float* __restrict__ bb) {
  unsigned short* p = data + (size_t)blockIdx.x * rowStride + segOff;
  int t = threadIdx.x;
  float x0 = bf2f(p[t]), x1 = bf2f(p[t + 256]), x2 = bf2f(p[t + 512]);
  __shared__ float r1[256], r2[256];
  r1[t] = x0 + x1 + x2;
  r2[t] = x0 * x0 + x1 * x1 + x2 * x2;
  __syncthreads();
  for (int o = 128; o > 0; o >>= 1) {
    if (t < o) { r1[t] += r1[t + o]; r2[t] += r2[t + o]; }
    __syncthreads();
  }
  float mean = r1[0] * (1.0f / 768.0f);
  float var  = r2[0] * (1.0f / 768.0f) - mean * mean;
  float rinv = rsqrtf(var + LN_EPS);
  p[t]       = f2bf((x0 - mean) * rinv * g[t]       + bb[t]);
  p[t + 256] = f2bf((x1 - mean) * rinv * g[t + 256] + bb[t + 256]);
  p[t + 512] = f2bf((x2 - mean) * rinv * g[t + 512] + bb[t + 512]);
}

// ---------------- in-place LayerNorm, f32 storage (full 768-row) ----------------
__global__ __launch_bounds__(256) void k_ln_f32(float* __restrict__ data,
                                                const float* __restrict__ g,
                                                const float* __restrict__ bb) {
  float* p = data + (size_t)blockIdx.x * 768;
  int t = threadIdx.x;
  float x0 = p[t], x1 = p[t + 256], x2 = p[t + 512];
  __shared__ float r1[256], r2[256];
  r1[t] = x0 + x1 + x2;
  r2[t] = x0 * x0 + x1 * x1 + x2 * x2;
  __syncthreads();
  for (int o = 128; o > 0; o >>= 1) {
    if (t < o) { r1[t] += r1[t + o]; r2[t] += r2[t + o]; }
    __syncthreads();
  }
  float mean = r1[0] * (1.0f / 768.0f);
  float var  = r2[0] * (1.0f / 768.0f) - mean * mean;
  float rinv = rsqrtf(var + LN_EPS);
  p[t]       = (x0 - mean) * rinv * g[t]       + bb[t];
  p[t + 256] = (x1 - mean) * rinv * g[t + 256] + bb[t + 256];
  p[t + 512] = (x2 - mean) * rinv * g[t + 512] + bb[t + 512];
}

// ---------------- instruction-feature attention: 64 queries x 77 keys ----------------
__global__ __launch_bounds__(256) void k_attn_inst(const float* __restrict__ queries,
                                                   const unsigned short* __restrict__ kvinst,
                                                   float* __restrict__ qa) {
  int bn = blockIdx.x, b = bn >> 6, n = bn & 63;
  __shared__ float q[768];
  __shared__ float sc[80];
  __shared__ float sinv;
  for (int i = threadIdx.x; i < 768; i += 256) q[i] = queries[n * 768 + i];
  __syncthreads();
  int wave = threadIdx.x >> 5, lane = threadIdx.x & 31;
  for (int m = wave; m < 77; m += 8) {
    const unsigned short* krow = kvinst + (size_t)(b * 77 + m) * 1536;
    float acc = 0.f;
    for (int d = lane; d < 768; d += 32) acc += q[d] * bf2f(krow[d]);
    for (int o = 16; o > 0; o >>= 1) acc += __shfl_xor(acc, o, 32);
    if (lane == 0) sc[m] = acc * QK_SCALE;
  }
  __syncthreads();
  if (threadIdx.x == 0) {
    float mx = -1e30f;
    for (int m = 0; m < 77; ++m) mx = fmaxf(mx, sc[m]);
    float sm = 0.f;
    for (int m = 0; m < 77; ++m) { float e = __expf(sc[m] - mx); sc[m] = e; sm += e; }
    sinv = 1.0f / sm;
  }
  __syncthreads();
  float inv = sinv;
#pragma unroll
  for (int i = 0; i < 3; ++i) {
    int d = threadIdx.x + i * 256;
    float acc = 0.f;
    for (int m = 0; m < 77; ++m)
      acc += sc[m] * bf2f(kvinst[(size_t)(b * 77 + m) * 1536 + 768 + d]);
    qa[(size_t)bn * 768 + d] = acc * inv;
  }
}

// ---------------- windowed multimodal attention: 64 queries x 512-key windows ----------------
__global__ __launch_bounds__(256) void k_attn_mm(const float* __restrict__ q2,
                                                 const unsigned short* __restrict__ kvmm,
                                                 float* __restrict__ ob) {
  int bn = blockIdx.x, b = bn >> 6, n = bn & 63;
  int s = n * 121;                         // window start (step=(8192-512)//63=121, never clipped)
  __shared__ float q[768];
  __shared__ float sc[512];
  __shared__ float red[256];
  int t = threadIdx.x;
  for (int i = t; i < 768; i += 256) q[i] = q2[(size_t)bn * 768 + i];
  __syncthreads();
  int wave = t >> 5, lane = t & 31;
  for (int j = 0; j < 64; ++j) {
    int w = wave * 64 + j;
    const unsigned short* krow = kvmm + (size_t)(b * 8192 + s + w) * 1536;
    float acc = 0.f;
    for (int d = lane; d < 768; d += 32) acc += q[d] * bf2f(krow[d]);
    for (int o = 16; o > 0; o >>= 1) acc += __shfl_xor(acc, o, 32);
    if (lane == 0) sc[w] = acc * QK_SCALE;
  }
  __syncthreads();
  red[t] = fmaxf(sc[t], sc[t + 256]);
  __syncthreads();
  for (int o = 128; o > 0; o >>= 1) {
    if (t < o) red[t] = fmaxf(red[t], red[t + o]);
    __syncthreads();
  }
  float mx = red[0];
  __syncthreads();
  float e0 = __expf(sc[t] - mx), e1 = __expf(sc[t + 256] - mx);
  sc[t] = e0; sc[t + 256] = e1;
  red[t] = e0 + e1;
  __syncthreads();
  for (int o = 128; o > 0; o >>= 1) {
    if (t < o) red[t] += red[t + o];
    __syncthreads();
  }
  float inv = 1.0f / red[0];
#pragma unroll
  for (int i = 0; i < 3; ++i) {
    int d = t + i * 256;
    float acc = 0.f;
    for (int w = 0; w < 512; ++w)
      acc += sc[w] * bf2f(kvmm[(size_t)(b * 8192 + s + w) * 1536 + 768 + d]);
    ob[(size_t)bn * 768 + d] = acc * inv;
  }
}

extern "C" void kernel_launch(void* const* d_in, const int* in_sizes, int n_in,
                              void* d_out, int out_size, void* d_ws, size_t ws_size,
                              hipStream_t stream) {
  const float* x         = (const float*)d_in[0];   // 4x8192x1024
  const float* instf     = (const float*)d_in[1];   // 4x77x768
  const float* queries   = (const float*)d_in[2];   // 1x64x768
  const float* W_proj    = (const float*)d_in[3];
  const float* b_proj    = (const float*)d_in[4];
  const float* W_kv_inst = (const float*)d_in[5];
  const float* b_kv_inst = (const float*)d_in[6];
  const float* W_out_inst= (const float*)d_in[7];
  const float* b_out_inst= (const float*)d_in[8];
  const float* W_kv_mm   = (const float*)d_in[9];
  const float* b_kv_mm   = (const float*)d_in[10];
  const float* W_out_mm  = (const float*)d_in[11];
  const float* b_out_mm  = (const float*)d_in[12];
  const float* ln_g      = (const float*)d_in[13];
  const float* ln_b      = (const float*)d_in[14];
  float* out = (float*)d_out;

  char* w = (char*)d_ws;
  auto alloc = [&](size_t bytes) { char* p = w; w += bytes; return p; };
  unsigned short* xb        = (unsigned short*)alloc(32768ull * 1024 * 2);  // x bf16
  unsigned short* WkvmmT    = (unsigned short*)alloc(1536ull * 1024 * 2);
  unsigned short* kvmm      = (unsigned short*)alloc(32768ull * 1536 * 2);  // k|v, LN'd in place
  unsigned short* instf_b   = (unsigned short*)alloc(320ull * 768 * 2);     // padded to 320 rows
  unsigned short* WprojT    = (unsigned short*)alloc(768ull * 768 * 2);
  unsigned short* inst_b    = (unsigned short*)alloc(320ull * 768 * 2);
  unsigned short* WkvinstT  = (unsigned short*)alloc(1536ull * 768 * 2);
  unsigned short* kvinst    = (unsigned short*)alloc(320ull * 1536 * 2);
  float*          qa        = (float*)alloc(256ull * 768 * 4);
  unsigned short* qa_b      = (unsigned short*)alloc(256ull * 768 * 2);
  unsigned short* WoutinstT = (unsigned short*)alloc(768ull * 768 * 2);
  float*          q2        = (float*)alloc(256ull * 768 * 4);
  float*          ob        = (float*)alloc(256ull * 768 * 4);
  unsigned short* ob_b      = (unsigned short*)alloc(256ull * 768 * 2);
  unsigned short* WoutmmT   = (unsigned short*)alloc(768ull * 768 * 2);

  // ---- multimodal K/V branch (the 103 GFLOP GEMM) ----
  k_cvt_bf16<<<(33554432 + 255) / 256, 256, 0, stream>>>(x, xb, 33554432);
  k_transpose_cvt<<<(1572864 + 255) / 256, 256, 0, stream>>>(W_kv_mm, WkvmmT, 1024, 1536);
  k_gemm_bf16<true><<<dim3(6, 1024), 256, 0, stream>>>(xb, WkvmmT, b_kv_mm, kvmm, 32768, 1024);
  k_ln_bf16<<<32768, 256, 0, stream>>>(kvmm, 1536, 0,   ln_g, ln_b);
  k_ln_bf16<<<32768, 256, 0, stream>>>(kvmm, 1536, 768, ln_g, ln_b);

  // ---- instruction branch ----
  k_cvt_bf16<<<(236544 + 255) / 256, 256, 0, stream>>>(instf, instf_b, 236544);
  k_transpose_cvt<<<(589824 + 255) / 256, 256, 0, stream>>>(W_proj, WprojT, 768, 768);
  k_gemm_bf16<true><<<dim3(3, 10), 256, 0, stream>>>(instf_b, WprojT, b_proj, inst_b, 320, 768);
  k_transpose_cvt<<<(1179648 + 255) / 256, 256, 0, stream>>>(W_kv_inst, WkvinstT, 768, 1536);
  k_gemm_bf16<true><<<dim3(6, 10), 256, 0, stream>>>(inst_b, WkvinstT, b_kv_inst, kvinst, 320, 768);
  k_ln_bf16<<<308, 256, 0, stream>>>(kvinst, 1536, 0,   ln_g, ln_b);
  k_ln_bf16<<<308, 256, 0, stream>>>(kvinst, 1536, 768, ln_g, ln_b);
  k_attn_inst<<<256, 256, 0, stream>>>(queries, kvinst, qa);
  k_cvt_bf16<<<(196608 + 255) / 256, 256, 0, stream>>>(qa, qa_b, 196608);
  k_transpose_cvt<<<(589824 + 255) / 256, 256, 0, stream>>>(W_out_inst, WoutinstT, 768, 768);
  k_gemm_bf16<false><<<dim3(3, 8), 256, 0, stream>>>(qa_b, WoutinstT, b_out_inst, q2, 256, 768);
  k_ln_f32<<<256, 256, 0, stream>>>(q2, ln_g, ln_b);

  // ---- windowed attention + output projection ----
  k_attn_mm<<<256, 256, 0, stream>>>(q2, kvmm, ob);
  k_cvt_bf16<<<(196608 + 255) / 256, 256, 0, stream>>>(ob, ob_b, 196608);
  k_transpose_cvt<<<(589824 + 255) / 256, 256, 0, stream>>>(W_out_mm, WoutmmT, 768, 768);
  k_gemm_bf16<false><<<dim3(3, 8), 256, 0, stream>>>(ob_b, WoutmmT, b_out_mm, out, 256, 768);
  k_ln_f32<<<256, 256, 0, stream>>>(out, ln_g, ln_b);
}
